// GRURNN_34686155882626
// MI455X (gfx1250) — compile-verified, matrix operand order checked by reading, not run on hardware
//
#include <hip/hip_runtime.h>
#include <hip/hip_bf16.h>

typedef __bf16 bf16;
typedef __attribute__((ext_vector_type(16))) __bf16 v16bf;
typedef __attribute__((ext_vector_type(8)))  float  v8f;
typedef __attribute__((ext_vector_type(8)))  short  short8;
typedef __attribute__((ext_vector_type(4)))  int    v4i;

// address-space-qualified element types for the async-to-LDS builtin
typedef __attribute__((address_space(1))) v4i gv4i;   // global int4
typedef __attribute__((address_space(3))) v4i sv4i;   // LDS int4

#define B_SZ   16384
#define IN_SZ  1024
#define H_SZ   1024
#define OUT_SZ 1024
#define KGATE  2048   // IN + H

#define BM 128
#define BN 128
#define BK 64

#if defined(__AMDGCN__) && __has_builtin(__builtin_amdgcn_global_load_async_to_lds_b128)
#define USE_ASYNC 1
#else
#define USE_ASYNC 0
#endif

// ---------- helpers ----------
__device__ __forceinline__ bf16 f2bf(float f) {
  unsigned u = __builtin_bit_cast(unsigned, f);
  unsigned r = u + 0x7FFFu + ((u >> 16) & 1u);   // RNE
  unsigned short hv = (unsigned short)(r >> 16);
  return __builtin_bit_cast(bf16, hv);
}

__device__ __forceinline__ float sigmoidf_(float x) {
  return 1.0f / (1.0f + __expf(-x));
}

template <int N>
__device__ __forceinline__ void wait_async() {
#if USE_ASYNC
#if __has_builtin(__builtin_amdgcn_s_wait_asynccnt)
  __builtin_amdgcn_s_wait_asynccnt((unsigned short)N);
#else
  asm volatile("s_wait_asynccnt %0" ::"i"(N) : "memory");
#endif
#endif
}

// Stage one 128x64 bf16 tile (16 KB) from global into LDS.
// 1024 16-byte chunks; 256 threads -> 4 chunks each.
__device__ __forceinline__ void tile_to_lds(const bf16* __restrict__ gbase, int ld,
                                            int row0, int k0, bf16* lds, int t) {
#pragma unroll
  for (int j = 0; j < 4; ++j) {
    const int c  = t + 256 * j;          // chunk id
    const int r  = c >> 3;               // tile row (0..127)
    const int ck = (c & 7) * 8;          // bf16 col offset (16B chunks)
    const bf16* g = gbase + (size_t)(row0 + r) * ld + k0 + ck;
    bf16* l = lds + r * BK + ck;
#if USE_ASYNC
    const v4i* gg = reinterpret_cast<const v4i*>(g);
    v4i*       ll = reinterpret_cast<v4i*>(l);
    __builtin_amdgcn_global_load_async_to_lds_b128((gv4i*)gg, (sv4i*)ll, 0, 0);
#else
    *reinterpret_cast<short8*>(l) = *reinterpret_cast<const short8*>(g);
#endif
  }
}

// Read one 16x32 bf16 WMMA operand fragment from an LDS tile (row-major, ld=BK).
// lane<16 : row=r16+lane,    K={kh..kh+7, kh+16..kh+23}
// lane>=16: row=r16+lane-16, K shifted by 8
__device__ __forceinline__ v16bf lds_frag(const bf16* lds, int r16, int ksub, int lane) {
  const int r  = r16 + (lane & 15);
  const int kh = ksub * 32 + ((lane >> 4) << 3);
  const bf16* p = lds + r * BK + kh;
  union { short8 s[2]; v16bf v; } u;
  u.s[0] = *reinterpret_cast<const short8*>(p);
  u.s[1] = *reinterpret_cast<const short8*>(p + 16);
  return u.v;
}

// ---------- conversion / packing kernels ----------
__global__ __launch_bounds__(256) void cvt_kernel(const float* __restrict__ src,
                                                  bf16* __restrict__ dst, int n) {
  int i = blockIdx.x * 256 + threadIdx.x;
  if (i < n) dst[i] = f2bf(src[i]);
}

__global__ __launch_bounds__(256) void pack_xh_kernel(const float* __restrict__ x,
                                                      const float* __restrict__ h,
                                                      bf16* __restrict__ xh,
                                                      bf16* __restrict__ xrh) {
  size_t i = (size_t)blockIdx.x * 256 + threadIdx.x;     // over B*IN
  if (i >= (size_t)B_SZ * IN_SZ) return;
  size_t row = i >> 10;
  size_t col = i & 1023;
  bf16 xb = f2bf(x[i]);
  xh [row * KGATE + col]         = xb;
  xh [row * KGATE + IN_SZ + col] = f2bf(h[i]);
  xrh[row * KGATE + col]         = xb;     // h-half of xrh filled by GEMM1 epilogue
}

// ---------- fused WMMA GEMM: D = A[M,K] @ W[N,K]^T (+ epilogue) ----------
// Block: 256 threads = 8 waves (4 down x 2 across), 128x128 tile,
// double-buffered async LDS staging of 128x64 A/B slices.
// MODE 0: gates.   col<1024: reset -> xrh h-half (bf16). else: update -> u_buf (f32)
// MODE 1: cand.    tanh + blend -> new_hidden (f32 out) + nh bf16 copy
// MODE 2: output.  +bias -> f32 out
template <int MODE>
__global__ __launch_bounds__(256)
void gru_gemm(const bf16* __restrict__ A, const bf16* __restrict__ W, int K,
              const float* __restrict__ bias0, const float* __restrict__ bias1,
              const float* __restrict__ hidden,
              float* __restrict__ f32buf,    // MODE0: u out ; MODE1: u in
              bf16*  __restrict__ bf16out,   // MODE0: xrh ; MODE1: nh_bf
              float* __restrict__ f32out)    // MODE1: new_hidden ; MODE2: output
{
  __shared__ bf16 Ash[2][BM * BK];   // 2 x 16 KB
  __shared__ bf16 Bsh[2][BN * BK];   // 2 x 16 KB

  const int t    = threadIdx.x;
  const int lane = t & 31;
  const int wid  = t >> 5;
  const int wm   = wid & 3;                       // 4 waves down
  const int wcol = wid >> 2;                      // 2 waves across
  const int rowW = wm * 32;                       // wave rows within block tile
  const int colW = wcol * 64;                     // wave cols within block tile
  const int blockRow = blockIdx.y * BM;
  const int blockCol = blockIdx.x * BN;

  v8f acc[2][4];
  const v8f vzero = {0.f, 0.f, 0.f, 0.f, 0.f, 0.f, 0.f, 0.f};
#pragma unroll
  for (int tm = 0; tm < 2; ++tm)
#pragma unroll
    for (int tn = 0; tn < 4; ++tn) acc[tm][tn] = vzero;

  // prologue: fill buffer 0
  tile_to_lds(A, K, blockRow, 0, Ash[0], t);
  tile_to_lds(W, K, blockCol, 0, Bsh[0], t);

  const int nIter = K / BK;
  for (int it = 0; it < nIter; ++it) {
    const int cur = it & 1;
    if (it + 1 < nIter) {
      // prefetch next K-slice into the other buffer (8 async ops per wave)
      tile_to_lds(A, K, blockRow, (it + 1) * BK, Ash[cur ^ 1], t);
      tile_to_lds(W, K, blockCol, (it + 1) * BK, Bsh[cur ^ 1], t);
      wait_async<8>();      // in-order completion => current buffer resident
    } else {
      wait_async<0>();
    }
    __syncthreads();

#pragma unroll
    for (int ksub = 0; ksub < 2; ++ksub) {        // two 32-deep WMMA steps
      v16bf a[2], b[4];
#pragma unroll
      for (int tm = 0; tm < 2; ++tm)
        a[tm] = lds_frag(Ash[cur], rowW + tm * 16, ksub, lane);
#pragma unroll
      for (int tn = 0; tn < 4; ++tn)
        b[tn] = lds_frag(Bsh[cur], colW + tn * 16, ksub, lane);
#pragma unroll
      for (int tm = 0; tm < 2; ++tm)
#pragma unroll
        for (int tn = 0; tn < 4; ++tn)
          acc[tm][tn] = __builtin_amdgcn_wmma_f32_16x16x32_bf16(
              false, a[tm], false, b[tn], (short)0, acc[tm][tn], false, false);
    }
    __syncthreads();   // protect buffer before it is overwritten next iteration
  }

  // epilogue: C/D layout -> lane<16: M=i, N=lane ; lane>=16: M=i+8, N=lane-16
  const int row0 = blockRow + rowW;
  const int col0 = blockCol + colW;
#pragma unroll
  for (int tm = 0; tm < 2; ++tm) {
#pragma unroll
    for (int tn = 0; tn < 4; ++tn) {
      const int n0    = col0 + tn * 16;
      const int col   = n0 + (lane & 15);
      const int rbase = row0 + tm * 16 + ((lane >> 4) << 3);
#pragma unroll
      for (int i = 0; i < 8; ++i) {
        const int row = rbase + i;
        const float v = acc[tm][tn][i];
        if constexpr (MODE == 0) {
          if (n0 < H_SZ) {                                   // reset gate
            float r  = sigmoidf_(v + bias0[col]);
            float rh = r * hidden[(size_t)row * H_SZ + col];
            bf16out[(size_t)row * KGATE + IN_SZ + col] = f2bf(rh);
          } else {                                           // update gate
            int uc = col - H_SZ;
            f32buf[(size_t)row * H_SZ + uc] = sigmoidf_(v + bias1[uc]);
          }
        } else if constexpr (MODE == 1) {
          size_t idx = (size_t)row * H_SZ + col;
          float cand = tanhf(v + bias0[col]);
          float uu   = f32buf[idx];
          float nh   = (1.0f - uu) * cand + uu * hidden[idx];
          f32out[idx]  = nh;
          bf16out[idx] = f2bf(nh);
        } else {
          f32out[(size_t)row * OUT_SZ + col] = v + bias0[col];
        }
      }
    }
  }
}

// ---------- launch ----------
extern "C" void kernel_launch(void* const* d_in, const int* in_sizes, int n_in,
                              void* d_out, int out_size, void* d_ws, size_t ws_size,
                              hipStream_t stream) {
  (void)in_sizes; (void)n_in; (void)out_size; (void)ws_size;
  const float* x      = (const float*)d_in[0];
  const float* hidden = (const float*)d_in[1];
  const float* W_r    = (const float*)d_in[2];
  const float* b_r    = (const float*)d_in[3];
  const float* W_z    = (const float*)d_in[4];
  const float* b_z    = (const float*)d_in[5];
  const float* W_c    = (const float*)d_in[6];
  const float* b_c    = (const float*)d_in[7];
  const float* W_o    = (const float*)d_in[8];
  const float* b_o    = (const float*)d_in[9];

  float* out    = (float*)d_out;                       // [B, OUT]
  float* out_nh = out + (size_t)B_SZ * OUT_SZ;         // [B, H]

  // workspace carve-up (all offsets 16B-aligned)
  bf16*  xh    = (bf16*)d_ws;                          // [B, 2048]  64 MB
  bf16*  xrh   = xh  + (size_t)B_SZ * KGATE;           // [B, 2048]  64 MB
  bf16*  wrz   = xrh + (size_t)B_SZ * KGATE;           // [2048,2048] 8 MB (W_r rows, then W_z)
  bf16*  wc    = wrz + (size_t)2 * H_SZ * KGATE;       // [1024,2048] 4 MB
  bf16*  wo    = wc  + (size_t)H_SZ * KGATE;           // [1024,1024] 2 MB
  float* u_buf = (float*)(wo + (size_t)OUT_SZ * H_SZ); // [B, H]     64 MB
  bf16*  nh_bf = (bf16*)(u_buf + (size_t)B_SZ * H_SZ); // [B, H]     32 MB

  const int wn  = H_SZ * KGATE;                        // 2M elems per gate weight
  const int won = OUT_SZ * H_SZ;
  cvt_kernel<<<(wn  + 255) / 256, 256, 0, stream>>>(W_r, wrz,      wn);
  cvt_kernel<<<(wn  + 255) / 256, 256, 0, stream>>>(W_z, wrz + wn, wn);
  cvt_kernel<<<(wn  + 255) / 256, 256, 0, stream>>>(W_c, wc,       wn);
  cvt_kernel<<<(won + 255) / 256, 256, 0, stream>>>(W_o, wo,       won);

  const size_t pn = (size_t)B_SZ * IN_SZ;
  pack_xh_kernel<<<(unsigned)((pn + 255) / 256), 256, 0, stream>>>(x, hidden, xh, xrh);

  // GEMM1: gates  [B,2048] = xh @ [W_r;W_z]^T   (K=2048)
  gru_gemm<0><<<dim3(2048 / BN, B_SZ / BM), 256, 0, stream>>>(
      xh, wrz, KGATE, b_r, b_z, hidden, u_buf, xrh, nullptr);

  // GEMM2: candidate + blend  [B,1024] = [x|rh] @ W_c^T   (K=2048)
  gru_gemm<1><<<dim3(1024 / BN, B_SZ / BM), 256, 0, stream>>>(
      xrh, wc, KGATE, b_c, nullptr, hidden, u_buf, nh_bf, out_nh);

  // GEMM3: output  [B,1024] = nh @ W_o^T + b_o   (K=1024)
  gru_gemm<2><<<dim3(1024 / BN, B_SZ / BM), 256, 0, stream>>>(
      nh_bf, wo, H_SZ, b_o, nullptr, nullptr, nullptr, nullptr, out);
}
